// RBFGIMutilAdaptiveDecoder_82489141887084
// MI455X (gfx1250) — matrix operand order repaired and checked by reading, not compiled, not run on hardware
//
#include <hip/hip_runtime.h>
#include <hip/hip_bf16.h>

// Problem dims
#define H_IMG   256
#define W_IMG   256
#define NPIX    (H_IMG * W_IMG)        // 65536  (GEMM N)
#define NCP     164
#define NC      3
#define KDIM    492                     // 3 * 164
#define KPAD    512                     // padded inner dim
#define BD      64
#define KIW     10
#define NJ      1280                    // K*B*2   (GEMM M)
#define ALPHA_ELEMS (KIW * NC * BD * NCP * 2)      // 629760
#define PHI_ELEMS   ((long)KIW * BD * 2 * NPIX)    // 83886080

typedef __attribute__((ext_vector_type(16))) _Float16 v16h;
typedef __attribute__((ext_vector_type(8)))  _Float16 v8h;
typedef __attribute__((ext_vector_type(8)))  float    v8f;
typedef __attribute__((ext_vector_type(4)))  int      v4i;

// ---------------------------------------------------------------------------
// CDNA5 async global->LDS copy (ASYNCcnt path), with sync fallback.
// Probe result: builtin exists; params are (int4 AS1*, int4 AS3*, imm, imm).
// ---------------------------------------------------------------------------
#if __has_builtin(__builtin_amdgcn_global_load_async_to_lds_b128)
#define HAVE_ASYNC_LDS 1
#else
#define HAVE_ASYNC_LDS 0
#warning "gfx1250 async-to-LDS builtin NOT available; using sync load+ds_store fallback"
#endif

typedef v4i __attribute__((address_space(1)))* gv4i_p;   // global int4*
typedef v4i __attribute__((address_space(3)))* lv4i_p;   // LDS int4*

__device__ __forceinline__ void cp16_async(_Float16* lds, const _Float16* g) {
#if HAVE_ASYNC_LDS
    __builtin_amdgcn_global_load_async_to_lds_b128(
        (gv4i_p)(v4i*)g,        // strip const, retype, addrspacecast -> AS1
        (lv4i_p)(v4i*)lds,      // addrspacecast -> AS3 (raw LDS offset)
        0 /*imm offset*/, 0 /*cpol*/);
#else
    *(v8h*)lds = *(const v8h*)g;
#endif
}

__device__ __forceinline__ void wait_async0() {
#if __has_builtin(__builtin_amdgcn_s_wait_asynccnt)
    __builtin_amdgcn_s_wait_asynccnt(0);
#elif HAVE_ASYNC_LDS
    asm volatile("s_wait_asynccnt 0x0" ::: "memory");
#endif
}

// ---------------------------------------------------------------------------
// K1: alpha = mu + exp(0.5*var)*eps ; write fp32 alpha to d_out, f16 A-operand
//     matrix Aws[j=1280][i=512] to workspace, and the scale tail of d_out.
// ---------------------------------------------------------------------------
__global__ void alpha_kernel(const float* __restrict__ mu,
                             const float* __restrict__ var,
                             const float* __restrict__ eps,
                             const float* __restrict__ scale,
                             float* __restrict__ out,
                             _Float16* __restrict__ Aws) {
    const int gid = blockIdx.x * 256 + threadIdx.x;   // 1280*512 threads
    const int j = gid >> 9;          // 0..1279  == ((k*64 + b)*2 + d)
    const int i = gid & 511;         // 0..511   == c*164 + n (padded)
    _Float16 hv = (_Float16)0.0f;
    if (i < KDIM) {
        const int c = i / NCP;
        const int n = i - c * NCP;
        const int d = j & 1;
        const int b = (j >> 1) & 63;
        const int k = j >> 7;
        const long mi = (((long)(c * BD + b)) * NCP + n) * 2 + d;
        const long ei = (((long)((k * NC + c) * BD + b)) * NCP + n) * 2 + d;
        const float a = mu[mi] + __expf(0.5f * var[mi]) * eps[ei];
        out[ei] = a;                 // alpha output (first chunk of d_out)
        hv = (_Float16)a;
    }
    Aws[(long)j * KPAD + i] = hv;
    if (gid == 0) {                  // scale passthrough (last 2 outputs)
        out[ALPHA_ELEMS + PHI_ELEMS + 0] = scale[0];
        out[ALPHA_ELEMS + PHI_ELEMS + 1] = scale[1];
    }
}

// ---------------------------------------------------------------------------
// K2: materialize Wendland RBF weights as f16, layout Wws[p=65536][i=512]
//     (B-operand, k-contiguous per pixel). 64 MB -> lives in L2 (192 MB).
// ---------------------------------------------------------------------------
__global__ void wgen_kernel(const float* __restrict__ scale,
                            _Float16* __restrict__ Wws) {
    const int gid = blockIdx.x * 256 + threadIdx.x;   // 65536*64 threads
    const int p  = gid >> 6;
    const int i0 = (gid & 63) * 8;
    const float x = (float)(p & 255) * (8.0f / 255.0f);
    const float y = (float)(p >> 8)  * (8.0f / 255.0f);
    const float s0 = scale[0], s1 = scale[1];
    v8h outv;
#pragma unroll
    for (int e = 0; e < 8; ++e) {
        const int i = i0 + e;
        float w = 0.0f;
        if (i < KDIM) {
            const int c = i / NCP;
            const int n = i - c * NCP;
            float cx, cy;
            if (n < 64) {                         // global 8x8 grid, step 1.0
                cx = 0.5f + (float)(n & 7);
                cy = 0.5f + (float)(n >> 3);
            } else {                              // local 10x10 grid, scaled
                const int m  = n - 64;
                const int jj = m % 10;
                const int ii = m / 10;
                cx = (1.8f + (float)jj * (4.4f / 9.0f) - 4.0f) * s0 + 4.0f;
                cy = (1.8f + (float)ii * (4.4f / 9.0f) - 4.0f) * s1 + 4.0f;
            }
            const float cinv = (c == 0) ? (1.0f / 1.5f) : ((c == 1) ? 0.5f : 0.4f);
            const float dx = x - cx, dy = y - cy;
            const float dd = __fsqrt_rn(dx * dx + dy * dy) * cinv;
            if (dd < 1.0f) {
                float t = 1.0f - dd;
                t = t * t;  t = t * t;            // (1-d)^4
                w = t * (4.0f * dd + 1.0f);
            }
        }
        outv[e] = (_Float16)w;
    }
    *(v8h*)(Wws + (long)p * KPAD + i0) = outv;
}

// ---------------------------------------------------------------------------
// K3: WMMA GEMM  phi[j][p] = sum_i A[j][i] * W[i][p]
//     256 threads = 8 waves (wave32); block tile 64(j) x 128(p);
//     wave (wm, wn) computes 16(j) x 64(p) as 4 fp32 accumulators.
//     Double-buffered LDS, async global->LDS staging, 1 barrier / k-step.
// ---------------------------------------------------------------------------
__global__ void __launch_bounds__(256)
gemm_kernel(const _Float16* __restrict__ Aws,   // [1280][512]
            const _Float16* __restrict__ Wws,   // [65536][512]
            float* __restrict__ phi) {          // [1280][65536]
    __shared__ _Float16 Atile[2][64 * 32];    // [buf][row j][k]   2 x 4 KB
    __shared__ _Float16 Btile[2][128 * 32];   // [buf][col p][k]   2 x 8 KB

    const int tid  = threadIdx.x;
    const int lane = tid & 31;
    const int wave = tid >> 5;
    const int wm   = wave & 3;             // j strip within block (16 rows)
    const int wn   = wave >> 2;            // p strip within block (64 cols)
    const long jblk = (long)blockIdx.y * 64;
    const long pblk = (long)blockIdx.x * 128;

    // per-thread staging slots (16 B each): 1 for A tile, 2 for B tile
    const int ar   = tid >> 2;             // A row 0..63
    const int ach  = (tid & 3) * 8;        // A k-chunk
    const int bp0  = tid >> 2;             // B col 0..63
    const int bch0 = (tid & 3) * 8;
    const int bp1  = (tid + 256) >> 2;     // B col 64..127
    const int bch1 = bch0;
    const _Float16* ag  = Aws + (jblk + ar)  * KPAD + ach;
    const _Float16* bg0 = Wws + (pblk + bp0) * KPAD + bch0;
    const _Float16* bg1 = Wws + (pblk + bp1) * KPAD + bch1;

    v8f acc[4] = {};                        // 4 x 16x16 fp32 accumulators

    const int hihalf = (lane & 16) ? 1 : 0;
    const int l15    = lane & 15;

    // prologue: stage K-slice 0 into buffer 0
    cp16_async(&Atile[0][ar * 32 + ach],   ag);
    cp16_async(&Btile[0][bp0 * 32 + bch0], bg0);
    cp16_async(&Btile[0][bp1 * 32 + bch1], bg1);

#pragma unroll 1
    for (int it = 0; it < KPAD / 32; ++it) {
        const int buf = it & 1;
        wait_async0();                     // my stage of buf complete
        __syncthreads();                   // everyone staged buf; prev reads done

        if (it + 1 < KPAD / 32) {          // async-stage next slice (other buf)
            const int kk = (it + 1) * 32;
            cp16_async(&Atile[buf ^ 1][ar * 32 + ach],   ag  + kk);
            cp16_async(&Btile[buf ^ 1][bp0 * 32 + bch0], bg0 + kk);
            cp16_async(&Btile[buf ^ 1][bp1 * 32 + bch1], bg1 + kk);
        }

        // ---- A fragment (16x32 f16): lane = row; K 0..7/16..23 (lanes 0-15)
        //      vs 8..15/24..31 (lanes 16-31), per ISA 7.12.2 ----
        const int arow = wm * 16 + l15;
        const int klo  = hihalf ? 8 : 0;
        const v8h alo = *(const v8h*)&Atile[buf][arow * 32 + klo];
        const v8h ahi = *(const v8h*)&Atile[buf][arow * 32 + klo + 16];
        const v16h afrag = __builtin_shufflevector(
            alo, ahi, 0, 1, 2, 3, 4, 5, 6, 7, 8, 9, 10, 11, 12, 13, 14, 15);

#pragma unroll
        for (int tn = 0; tn < 4; ++tn) {
            // ---- B fragment (32x16 f16): lane = col; K 0..7/16..23 / 8..15/24..31 ----
            const int pl = wn * 64 + tn * 16 + l15;
            const int kb = hihalf ? 8 : 0;
            const v8h blo = *(const v8h*)&Btile[buf][pl * 32 + kb];
            const v8h bhi = *(const v8h*)&Btile[buf][pl * 32 + kb + 16];
            const v16h bfrag = __builtin_shufflevector(
                blo, bhi, 0, 1, 2, 3, 4, 5, 6, 7, 8, 9, 10, 11, 12, 13, 14, 15);
            acc[tn] = __builtin_amdgcn_wmma_f32_16x16x32_f16(
                false, afrag, false, bfrag, (short)0, acc[tn], false, false);
        }
    }

    // ---- store: C layout lane->N(p), VGPR v -> M(j) = v + 8*(lane>>4).
    //      Non-temporal: phi streams out once, keep L2 for the W operand. ----
    const long jrow0 = jblk + wm * 16 + (hihalf ? 8 : 0);
    const long pc    = pblk + wn * 64 + l15;
#pragma unroll
    for (int tn = 0; tn < 4; ++tn) {
#pragma unroll
        for (int v = 0; v < 8; ++v) {
            __builtin_nontemporal_store(
                acc[tn][v], &phi[(jrow0 + v) * (long)NPIX + pc + tn * 16]);
        }
    }
}

// ---------------------------------------------------------------------------
extern "C" void kernel_launch(void* const* d_in, const int* in_sizes, int n_in,
                              void* d_out, int out_size, void* d_ws, size_t ws_size,
                              hipStream_t stream) {
    const float* mu    = (const float*)d_in[0];
    const float* var   = (const float*)d_in[1];
    const float* eps   = (const float*)d_in[2];
    const float* scale = (const float*)d_in[3];
    float* out = (float*)d_out;

    _Float16* Aws = (_Float16*)d_ws;                                  // 1.31 MB
    _Float16* Wws = (_Float16*)((char*)d_ws + (size_t)NJ * KPAD * 2); // 64 MB

    // K1: alpha + A-operand (f16) + scale tail
    alpha_kernel<<<(NJ * KPAD) / 256, 256, 0, stream>>>(mu, var, eps, scale,
                                                        out, Aws);
    // K2: RBF weight matrix (f16), B-operand layout [p][i]
    wgen_kernel<<<((long)NPIX * 64) / 256, 256, 0, stream>>>(scale, Wws);
    // K3: WMMA GEMM -> phi
    gemm_kernel<<<dim3(NPIX / 128, NJ / 64), 256, 0, stream>>>(
        Aws, Wws, out + ALPHA_ELEMS);
}
